// EdgeGATLayer_53635551593089
// MI455X (gfx1250) — matrix operand order repaired and checked by reading, not compiled, not run on hardware
//
#include <hip/hip_runtime.h>
#include <hip/hip_bf16.h>

#define NNODES 50000
#define NEDGES 800000
#define DFEAT  64
#define KDIM   192      // 3 * DFEAT
#define KP     96       // bf16-pair dwords per row (192 bf16)
#define ROWSTR 100      // padded LDS row stride in dwords: 400B, 16B-aligned, conflict-free

typedef __attribute__((ext_vector_type(16))) __bf16        bf16x16;
typedef __attribute__((ext_vector_type(8)))  float         f32x8;
typedef __attribute__((ext_vector_type(8)))  unsigned int  u32x8;

// ---------- helpers ----------
__device__ __forceinline__ unsigned pack_bf16x2(float a, float b) {
  unsigned ua = __float_as_uint(a), ub = __float_as_uint(b);
  ua += 0x7FFFu + ((ua >> 16) & 1u);   // round-to-nearest-even
  ub += 0x7FFFu + ((ub >> 16) & 1u);
  return (ua >> 16) | (ub & 0xFFFF0000u);
}

// order-preserving float<->uint key for atomicMax-based segment max
__device__ __forceinline__ unsigned float_order_key(float f) {
  unsigned u = __float_as_uint(f);
  return (u & 0x80000000u) ? ~u : (u | 0x80000000u);
}
__device__ __forceinline__ float key_to_float(unsigned k) {
  return __uint_as_float((k & 0x80000000u) ? (k ^ 0x80000000u) : ~k);
}

// A fragment: dwords {0..3, 8..11} from p (16B aligned) -> two ds_load_b128
__device__ __forceinline__ bf16x16 ld_frag_gap(const unsigned* p) {
  const uint4* q = (const uint4*)p;
  uint4 x = q[0], y = q[2];
  u32x8 r;
  r[0] = x.x; r[1] = x.y; r[2] = x.z; r[3] = x.w;
  r[4] = y.x; r[5] = y.y; r[6] = y.z; r[7] = y.w;
  return __builtin_bit_cast(bf16x16, r);
}
// B fragment: dwords {0..7} from p (16B aligned) -> two ds_load_b128
__device__ __forceinline__ bf16x16 ld_frag_seq(const unsigned* p) {
  const uint4* q = (const uint4*)p;
  uint4 x = q[0], y = q[1];
  u32x8 r;
  r[0] = x.x; r[1] = x.y; r[2] = x.z; r[3] = x.w;
  r[4] = y.x; r[5] = y.y; r[6] = y.z; r[7] = y.w;
  return __builtin_bit_cast(bf16x16, r);
}

// ---------- kernel 0: zero workspace ----------
__global__ __launch_bounds__(256)
void init_ws_kernel(float* outAgg, float* inAgg, unsigned* smax, float* denom) {
  long long i = (long long)blockIdx.x * 256 + threadIdx.x;
  if (i < (long long)NNODES * DFEAT) { outAgg[i] = 0.f; inAgg[i] = 0.f; }
  if (i < NNODES) { smax[i] = 0u; denom[i] = 0.f; }
}

// ---------- kernel 1: attention logits + segment max ----------
__global__ __launch_bounds__(256)
void logits_kernel(const float* __restrict__ nodes, const int* __restrict__ senders,
                   const int* __restrict__ receivers, const float* __restrict__ Wa,
                   const float* __restrict__ ba, float* __restrict__ logit,
                   unsigned* __restrict__ smax) {
  long long e = (long long)blockIdx.x * 256 + threadIdx.x;
  if (e >= NEDGES) return;
  const int s = senders[e], r = receivers[e];
  const float4* wv = (const float4*)Wa;
  const float4* sn = (const float4*)(nodes + (long long)s * DFEAT);
  const float4* rn = (const float4*)(nodes + (long long)r * DFEAT);
  float acc = ba[0];
#pragma unroll
  for (int q = 0; q < 16; ++q) {
    float4 w = wv[q], x = sn[q];
    acc += w.x * x.x + w.y * x.y + w.z * x.z + w.w * x.w;
  }
#pragma unroll
  for (int q = 0; q < 16; ++q) {
    float4 w = wv[16 + q], x = rn[q];
    acc += w.x * x.x + w.y * x.y + w.z * x.z + w.w * x.w;
  }
  logit[e] = acc;
  atomicMax(smax + r, float_order_key(acc));
}

// ---------- kernel 2: exp + segment sum ----------
__global__ __launch_bounds__(256)
void softmax_kernel(const int* __restrict__ receivers, const float* __restrict__ logit,
                    const unsigned* __restrict__ smax, float* __restrict__ exv,
                    float* __restrict__ denom) {
  long long e = (long long)blockIdx.x * 256 + threadIdx.x;
  if (e >= NEDGES) return;
  const int r = receivers[e];
  const float m = key_to_float(smax[r]);
  const float x = __expf(logit[e] - m);
  exv[e] = x;
  atomicAdd(denom + r, x);
}

// ---------- shared WMMA core: one wave, two 16-row M-tiles, K=192 ----------
// aRow0: LDS row base for M-tile 0 rows (this lane's row), aRow1 for M-tile 1.
__device__ __forceinline__ void wmma_tiles2(const unsigned* __restrict__ aRow0,
                                            const unsigned* __restrict__ aRow1,
                                            const unsigned* __restrict__ sW,
                                            int half, int lm, f32x8 acc[2][4]) {
#pragma unroll
  for (int t = 0; t < 6; ++t) {
    // A 16x32 bf16 fragment: half0 holds K0-7 & K16-23, half1 holds K8-15 & K24-31
    bf16x16 a0 = ld_frag_gap(aRow0 + t * 16 + half * 4);
    bf16x16 a1 = ld_frag_gap(aRow1 + t * 16 + half * 4);
#pragma unroll
    for (int j = 0; j < 4; ++j) {
      // B 32x16 bf16 fragment: half0 holds K0-7 & K16-23, half1 holds K8-15 & K24-31 (col = lane%16)
      bf16x16 bfv = ld_frag_gap(sW + (j * 16 + lm) * ROWSTR + t * 16 + half * 4);
      acc[0][j] = __builtin_amdgcn_wmma_f32_16x16x32_bf16(
          false, a0, false, bfv, (short)0, acc[0][j], false, false);
      acc[1][j] = __builtin_amdgcn_wmma_f32_16x16x32_bf16(
          false, a1, false, bfv, (short)0, acc[1][j], false, false);
    }
  }
}

// dynamic LDS layout (dwords): sW[64*ROWSTR] | sA[128*ROWSTR] | sAtt[128]
#define SW_DW   (DFEAT * ROWSTR)
#define SA_DW   (128 * ROWSTR)
#define EDGE_LDS_BYTES ((SW_DW + SA_DW + 128) * 4)
#define NODE_LDS_BYTES ((SW_DW + SA_DW) * 4)

// ---------- kernel 3: edge GEMM + ReLU + attention + scatter ----------
__global__ __launch_bounds__(128)
void edge_gemm_kernel(const float* __restrict__ nodes, const float* __restrict__ edges,
                      const int* __restrict__ senders, const int* __restrict__ receivers,
                      const float* __restrict__ We, const float* __restrict__ be,
                      const float* __restrict__ exv, const float* __restrict__ denom,
                      float* __restrict__ wOut, float* __restrict__ outAgg,
                      float* __restrict__ inAgg) {
  extern __shared__ unsigned smem[];
  unsigned* sW   = smem;
  unsigned* sA   = smem + SW_DW;
  float*    sAtt = (float*)(smem + SW_DW + SA_DW);

  const int tid = threadIdx.x;
  const long long eBase = (long long)blockIdx.x * 128;

  // stage weights We[n][k] -> bf16 pairs, one row per output n
  for (int i = tid; i < DFEAT * KP; i += 128) {
    int n = i / KP, p = i - n * KP;
    const float* w = We + n * KDIM + p * 2;
    sW[n * ROWSTR + p] = pack_bf16x2(w[0], w[1]);
  }
  // stage A = concat(edge, sender, receiver) rows -> bf16
  for (int i = tid; i < 128 * KP; i += 128) {
    int e = i / KP, p = i - e * KP;
    long long ge = eBase + e;
    int f = p * 2;
    const float* src;
    if (f < DFEAT)            src = edges + ge * DFEAT + f;
    else if (f < 2 * DFEAT)   src = nodes + (long long)senders[ge]   * DFEAT + (f - DFEAT);
    else                      src = nodes + (long long)receivers[ge] * DFEAT + (f - 2 * DFEAT);
    sA[e * ROWSTR + p] = pack_bf16x2(src[0], src[1]);
  }
  {
    long long ge = eBase + tid;
    sAtt[tid] = exv[ge] / denom[receivers[ge]];
  }
  __syncthreads();

  const int wave = tid >> 5, lane = tid & 31;
  const int half = lane >> 4, lm = lane & 15;
  f32x8 acc[2][4] = {};

  const unsigned* aRow0 = sA + (wave * 32 + lm) * ROWSTR;
  wmma_tiles2(aRow0, aRow0 + 16 * ROWSTR, sW, half, lm, acc);

  // D layout: VGPR r -> M = r (lanes 0-15) / r+8 (lanes 16-31); N = j*16 + lane%16
#pragma unroll
  for (int mt = 0; mt < 2; ++mt) {
    const int rowBase = wave * 32 + mt * 16;
#pragma unroll
    for (int r = 0; r < 8; ++r) {
      const int m = half ? (r + 8) : r;
      const long long ge = eBase + rowBase + m;
      const float att = sAtt[rowBase + m];
      const long long sOff = (long long)senders[ge]   * DFEAT;
      const long long rOff = (long long)receivers[ge] * DFEAT;
#pragma unroll
      for (int j = 0; j < 4; ++j) {
        const int n = j * 16 + lm;
        float v = acc[mt][j][r] + be[n];
        v = fmaxf(v, 0.f) * att;
        wOut[ge * DFEAT + n] = v;
        atomicAdd(outAgg + sOff + n, v);
        atomicAdd(inAgg + rOff + n, v);
      }
    }
  }
}

// ---------- kernel 4: node GEMM + ReLU ----------
__global__ __launch_bounds__(128)
void node_gemm_kernel(const float* __restrict__ nodes, const float* __restrict__ outAgg,
                      const float* __restrict__ inAgg, const float* __restrict__ Wn,
                      const float* __restrict__ bn, float* __restrict__ nodeOut) {
  extern __shared__ unsigned smem[];
  unsigned* sW = smem;
  unsigned* sA = smem + SW_DW;

  const int tid = threadIdx.x;
  const long long gBase = (long long)blockIdx.x * 128;

  for (int i = tid; i < DFEAT * KP; i += 128) {
    int n = i / KP, p = i - n * KP;
    const float* w = Wn + n * KDIM + p * 2;
    sW[n * ROWSTR + p] = pack_bf16x2(w[0], w[1]);
  }
  for (int i = tid; i < 128 * KP; i += 128) {
    int e = i / KP, p = i - e * KP;
    long long g = gBase + e;
    float f0 = 0.f, f1 = 0.f;
    if (g < NNODES) {
      int f = p * 2;
      const float* src;
      if (f < DFEAT)          src = nodes  + g * DFEAT + f;
      else if (f < 2 * DFEAT) src = outAgg + g * DFEAT + (f - DFEAT);
      else                    src = inAgg  + g * DFEAT + (f - 2 * DFEAT);
      f0 = src[0]; f1 = src[1];
    }
    sA[e * ROWSTR + p] = pack_bf16x2(f0, f1);
  }
  __syncthreads();

  const int wave = tid >> 5, lane = tid & 31;
  const int half = lane >> 4, lm = lane & 15;
  f32x8 acc[2][4] = {};

  const unsigned* aRow0 = sA + (wave * 32 + lm) * ROWSTR;
  wmma_tiles2(aRow0, aRow0 + 16 * ROWSTR, sW, half, lm, acc);

#pragma unroll
  for (int mt = 0; mt < 2; ++mt) {
    const int rowBase = wave * 32 + mt * 16;
#pragma unroll
    for (int r = 0; r < 8; ++r) {
      const int m = half ? (r + 8) : r;
      const long long g = gBase + rowBase + m;
      if (g < NNODES) {
#pragma unroll
        for (int j = 0; j < 4; ++j) {
          const int n = j * 16 + lm;
          float v = acc[mt][j][r] + bn[n];
          nodeOut[g * DFEAT + n] = fmaxf(v, 0.f);
        }
      }
    }
  }
}

// ---------- launch ----------
extern "C" void kernel_launch(void* const* d_in, const int* in_sizes, int n_in,
                              void* d_out, int out_size, void* d_ws, size_t ws_size,
                              hipStream_t stream) {
  (void)in_sizes; (void)n_in; (void)out_size; (void)ws_size;
  const float* nodes     = (const float*)d_in[0];
  const float* edges     = (const float*)d_in[1];
  const int*   senders   = (const int*)d_in[2];
  const int*   receivers = (const int*)d_in[3];
  const float* We        = (const float*)d_in[4];
  const float* be        = (const float*)d_in[5];
  const float* Wn        = (const float*)d_in[6];
  const float* bn        = (const float*)d_in[7];
  const float* Wa        = (const float*)d_in[8];
  const float* ba        = (const float*)d_in[9];

  float* newNodes = (float*)d_out;                                   // [N,64]
  float* wEdges   = (float*)d_out + (long long)NNODES * DFEAT;       // [E,64]

  char* ws = (char*)d_ws;
  float*    outAgg = (float*)ws;    ws += sizeof(float) * (long long)NNODES * DFEAT;
  float*    inAgg  = (float*)ws;    ws += sizeof(float) * (long long)NNODES * DFEAT;
  float*    logit  = (float*)ws;    ws += sizeof(float) * (long long)NEDGES;
  float*    exv    = (float*)ws;    ws += sizeof(float) * (long long)NEDGES;
  unsigned* smax   = (unsigned*)ws; ws += sizeof(unsigned) * (long long)NNODES;
  float*    denom  = (float*)ws;

  init_ws_kernel<<<(NNODES * DFEAT + 255) / 256, 256, 0, stream>>>(outAgg, inAgg, smax, denom);
  logits_kernel<<<(NEDGES + 255) / 256, 256, 0, stream>>>(nodes, senders, receivers, Wa, ba, logit, smax);
  softmax_kernel<<<(NEDGES + 255) / 256, 256, 0, stream>>>(receivers, logit, smax, exv, denom);
  edge_gemm_kernel<<<NEDGES / 128, 128, EDGE_LDS_BYTES, stream>>>(
      nodes, edges, senders, receivers, We, be, exv, denom, wEdges, outAgg, inAgg);
  node_gemm_kernel<<<(NNODES + 127) / 128, 128, NODE_LDS_BYTES, stream>>>(
      nodes, outAgg, inAgg, Wn, bn, newNodes);
}
